// GraphormerGraphAttnBias_12970801234641
// MI455X (gfx1250) — compile-verified, matrix operand order checked by reading, not run on hardware
//
#include <hip/hip_runtime.h>
#include <hip/hip_bf16.h>

typedef __bf16 bf16_t;
typedef bf16_t v16bf __attribute__((ext_vector_type(16)));
typedef float  v8f   __attribute__((ext_vector_type(8)));

namespace {
constexpr int kB   = 16;    // graphs
constexpr int kN   = 256;   // nodes
constexpr int kH   = 32;    // heads
constexpr int kD   = 5;     // multi-hop distance (== MHD)
constexpr int kF   = 3;     // edge features
constexpr int kNP1 = 257;   // N + 1
}

// Interior: one wave per 16-pair tile (fixed b,i ; 16 consecutive j).
// D(16x16 f32) = A(16x32 bf16: mean edge encodings) x B(32x16 bf16: W_d),
// accumulated over the 5 distances via the WMMA C operand; two WMMAs per
// distance cover the 32 heads.
__global__ __launch_bounds__(256) void graphormer_interior(
    const float* __restrict__ attn_bias,     // (B, 257, 257)
    const int*   __restrict__ spatial_pos,   // (B, N, N)
    const int*   __restrict__ input_edges,   // (B, N, N, D, F)
    const float* __restrict__ edge_enc_w,    // (1537, H)
    const float* __restrict__ spatial_enc_w, // (512, H)
    const float* __restrict__ edge_dis_w,    // (128*H*H) -> use first D*H*H
    float*       __restrict__ out)           // (B, H, 257, 257)
{
    __shared__ float xch[8][16][kH];   // per-wave 16-pair x 32-head transpose buffer

    const int lane = threadIdx.x & 31;
    const int wid  = threadIdx.x >> 5;
    const int hi   = lane >> 4;        // which lane-half
    const int lo   = lane & 15;

    // ---- Loop-invariant B fragments: W_d[K][n] in bf16 WMMA B layout ----
    // lane -> column n = lo + 16*t ; slot s -> K = s + 8*hi + (s>=8 ? 8 : 0)
    v16bf Bfrag[kD][2];
#pragma unroll
    for (int d = 0; d < kD; ++d) {
#pragma unroll
        for (int t = 0; t < 2; ++t) {
            v16bf fr;
#pragma unroll
            for (int s = 0; s < 16; ++s) {
                const int K = s + 8 * hi + ((s >= 8) ? 8 : 0);
                const int n = lo + 16 * t;
                fr[s] = (bf16_t)edge_dis_w[d * (kH * kH) + K * kH + n];
            }
            Bfrag[d][t] = fr;
        }
    }

    const int nTiles = kB * kN * (kN / 16);             // 65536
    const int gw = blockIdx.x * (blockDim.x >> 5) + wid;
    const int nw = gridDim.x * (blockDim.x >> 5);

    for (int tile = gw; tile < nTiles; tile += nw) {
        const int b   = tile >> 12;        // / (256*16)
        const int rem = tile & 4095;
        const int i   = rem >> 4;
        const int j0  = (rem & 15) << 4;

        const int jA = j0 + lo;            // A-matrix row (M = lo) for this lane

        v8f D0 = {0.f, 0.f, 0.f, 0.f, 0.f, 0.f, 0.f, 0.f};
        v8f D1 = D0;

#pragma unroll
        for (int d = 0; d < kD; ++d) {
            // mean over 3 features of edge_enc_w rows, sampled at this lane's K slots
            float acc[16];
#pragma unroll
            for (int s = 0; s < 16; ++s) acc[s] = 0.f;

            const long eb = ((((long)b * kN + i) * kN + jA) * kD + d) * kF;
#pragma unroll
            for (int f = 0; f < kF; ++f) {
                const int e = input_edges[eb + f];
                const float* row = edge_enc_w + (long)e * kH;
#pragma unroll
                for (int s = 0; s < 8; ++s) acc[s] += row[8 * hi + s];        // K = 8*hi + s
#pragma unroll
                for (int s = 0; s < 8; ++s) acc[8 + s] += row[16 + 8 * hi + s]; // K = 16 + 8*hi + s
            }

            v16bf a;
#pragma unroll
            for (int s = 0; s < 16; ++s) a[s] = (bf16_t)(acc[s] * (1.0f / 3.0f));

            D0 = __builtin_amdgcn_wmma_f32_16x16x32_bf16(
                    false, a, false, Bfrag[d][0], (short)0, D0, false, false);
            D1 = __builtin_amdgcn_wmma_f32_16x16x32_bf16(
                    false, a, false, Bfrag[d][1], (short)0, D1, false, false);
        }

        // ---- transpose through LDS: D layout is lane->head, VGPR->row ----
#pragma unroll
        for (int r = 0; r < 8; ++r) {
            const int mm = r + 8 * hi;     // pair row within tile
            xch[wid][mm][lo]      = D0[r];
            xch[wid][mm][lo + 16] = D1[r];
        }
        asm volatile("s_wait_dscnt 0" ::: "memory");   // DS in-order per wave; fence compiler

        // ---- coalesced writeback: each half-wave writes 16 consecutive j for one head ----
        const int j  = j0 + lo;
        const int sp = spatial_pos[((long)b * kN + i) * kN + j];
        int spc = (sp == 0) ? 1 : sp;
        spc = (spc > 1) ? (spc - 1) : spc;
        spc = (spc > 5) ? 5 : spc;
        const float inv = 1.0f / (float)spc;
        const float ab2 = 2.0f * attn_bias[((long)b * kNP1 + (i + 1)) * kNP1 + (j + 1)];
        const float* senc = spatial_enc_w + (long)sp * kH;

#pragma unroll
        for (int hh = 0; hh < 16; ++hh) {
            const int h = 2 * hh + hi;
            const float v = xch[wid][lo][h];
            const float o = ab2 + senc[h] + v * inv;
            out[(((long)(b * kH + h)) * kNP1 + (i + 1)) * kNP1 + (j + 1)] = o;
        }
    }
}

// Borders: row 0 (all columns) and column 0 (rows 1..N): 2*attn_bias + vdist_w[h]
__global__ __launch_bounds__(256) void graphormer_border(
    const float* __restrict__ attn_bias,
    const float* __restrict__ vdist_w,
    float*       __restrict__ out)
{
    const int tid = blockIdx.x * blockDim.x + threadIdx.x;
    const int total = kB * kH * kNP1;
    if (tid >= total) return;
    const int c  = tid % kNP1;
    const int bh = tid / kNP1;
    const int h  = bh % kH;
    const int b  = bh / kH;
    const float t = vdist_w[h];

    out[(((long)bh) * kNP1 + 0) * kNP1 + c] =
        2.0f * attn_bias[((long)b * kNP1 + 0) * kNP1 + c] + t;
    if (c != 0) {
        out[(((long)bh) * kNP1 + c) * kNP1 + 0] =
            2.0f * attn_bias[((long)b * kNP1 + c) * kNP1 + 0] + t;
    }
}

extern "C" void kernel_launch(void* const* d_in, const int* in_sizes, int n_in,
                              void* d_out, int out_size, void* d_ws, size_t ws_size,
                              hipStream_t stream) {
    // setup_inputs order:
    // 0 input_nodes (unused) | 1 attn_bias | 2 spatial_pos | 3 input_edges
    // 4 attn_edge_type (unused) | 5 edge_enc_w | 6 spatial_enc_w | 7 edge_dis_w | 8 vdist_w
    const float* attn_bias     = (const float*)d_in[1];
    const int*   spatial_pos   = (const int*)d_in[2];
    const int*   input_edges   = (const int*)d_in[3];
    const float* edge_enc_w    = (const float*)d_in[5];
    const float* spatial_enc_w = (const float*)d_in[6];
    const float* edge_dis_w    = (const float*)d_in[7];
    const float* vdist_w       = (const float*)d_in[8];
    float* out = (float*)d_out;

    hipLaunchKernelGGL(graphormer_interior, dim3(1024), dim3(256), 0, stream,
                       attn_bias, spatial_pos, input_edges,
                       edge_enc_w, spatial_enc_w, edge_dis_w, out);

    const int total = kB * kH * kNP1;
    hipLaunchKernelGGL(graphormer_border, dim3((total + 255) / 256), dim3(256), 0, stream,
                       attn_bias, vdist_w, out);
}
